// DensePICNN_54520314855831
// MI455X (gfx1250) — matrix operand order repaired and checked by reading, not compiled, not run on hardware
//
#include <hip/hip_runtime.h>

// ---------------------------------------------------------------------------
// Dense PICNN forward, fused, bf16 WMMA (gfx1250 / CDNA5, wave32).
// DIM=8, DIMC=8, DIMH=64, L=4, N=262144. 32 samples per wave (2x16 tiles
// sharing B fragments), 4 waves per block.
// ---------------------------------------------------------------------------

typedef __bf16 bf16_t;
typedef __attribute__((ext_vector_type(8)))  __bf16 v8bf;
typedef __attribute__((ext_vector_type(16))) __bf16 v16bf;
typedef __attribute__((ext_vector_type(8)))  float  v8f;
typedef __attribute__((ext_vector_type(8)))  unsigned int v8u;

#define DEV __device__ __forceinline__

__constant__ float kCZ0 = 0.36787944117144233f;  // exp(-1)
__constant__ float kCZL = 0.5413248546129181f;   // log(e-1)

#define LOG2E 1.4426950408889634f
#define LN2   0.6931471805599453f

// Branchless softplus: max(v,0) + ln2*log2(1+exp2(-|v|*log2e)).
// 1+exp2(-|v|) is in [1,2] -> raw v_log_f32/v_exp_f32 are exact enough,
// no guard branch, no range fixup.
DEV float sp_f(float v) {
  float e = __builtin_amdgcn_exp2f(-LOG2E * __builtin_fabsf(v));
  float l = __builtin_amdgcn_logf(1.f + e);
  return __builtin_fmaxf(v, 0.f) + LN2 * l;
}
// Branchless ELU (alpha=1).
DEV float elu_f(float v) {
  float e = __builtin_amdgcn_exp2f(LOG2E * v) - 1.f;
  return v > 0.f ? v : e;
}

// ---- bf16 arena layout (element offsets) ----------------------------------
#define OFF_WZ0   0        // 64x32  (K pad 8->32)
#define OFF_WC    2048     // 64x32
#define OFF_WX0   4096     // 64x32
#define OFF_WX1   6144
#define OFF_WX2   8192
#define OFF_WX3   10240    // 16x32  (1 row valid)
#define OFF_WCZ0  10752    // 64x64
#define OFF_WCZ1  14848    // 128x64
#define OFF_WCZ2  23040    // 192x64
#define OFF_WCZ3  35328    // 256x64
#define OFF_WCX0  51712    // 16x64 (8 rows valid)
#define OFF_WCX1  52736
#define OFF_WCX2  53760
#define OFF_WCX3  54784
#define OFF_WCC0  55808    // 64x64
#define OFF_WCC1  59904
#define OFF_WCC2  64000
#define OFF_SPZ1  68096    // 64x64   softplus(W)*1/64
#define OFF_SPZ2  72192    // 64x128  softplus(W)*1/128
#define OFF_SPZ3  80384    // 64x192  softplus(W)*1/192
#define OFF_SPZ4  92672    // 16x256  softplus(W)*1/256 (1 row valid)
#define BF_TOTAL  96768
// ---- f32 affine region (float offsets) ------------------------------------
// [0..63] s0  [64..127] t0  [128..191] sc  [192..255] tc
// [256..895] czbias (64,128,192,256 at 256/320/448/640)
// [896..927] cxbias (4x8)  [928..1119] ps (3x64)  [1120..1311] pt (3x64)
// [1312] fs  [1313] ft

// A-fragment (16x32 bf16) from row-major buffer, per ISA A layout.
DEV v16bf ldfragA(const bf16_t* base, int stride, int kBase, int lane) {
  int r = lane & 15, hi = (lane >> 4) & 1;
  const bf16_t* p = base + r * stride + kBase + hi * 8;
  v8bf lo = *(const v8bf*)(p);
  v8bf hv = *(const v8bf*)(p + 16);
  return __builtin_shufflevector(lo, hv, 0,1,2,3,4,5,6,7,8,9,10,11,12,13,14,15);
}
// B-fragment (32x16 bf16): lane n holds weight row n; same K order as A.
DEV v16bf ldfragB(const bf16_t* base, int stride, int kBase, int lane) {
  int r = lane & 15, hi = (lane >> 4) & 1;
  const bf16_t* p = base + r * stride + kBase + hi * 8;
  v8bf lo = *(const v8bf*)(p);
  v8bf hv = *(const v8bf*)(p + 16);
  return __builtin_shufflevector(lo, hv, 0,1,2,3,4,5,6,7,8,9,10,11,12,13,14,15);
}

DEV v8f wmma_acc(v16bf a, v16bf b, v8f c) {
  return __builtin_amdgcn_wmma_f32_16x16x32_bf16(
      false, a, false, b, (short)0, c, false, false);
}

// Packed bf16 elementwise multiply: 8x v_pk_mul_bf16 per 16-elem fragment.
DEV v16bf pkmul(v16bf a, v16bf b) {
  v8u ua = (v8u)a, ub = (v8u)b, ur;
#pragma unroll
  for (int i = 0; i < 8; ++i) {
    unsigned int x = ua[i], y = ub[i], r;
    asm("v_pk_mul_bf16 %0, %1, %2" : "=v"(r) : "v"(x), "v"(y));
    ur[i] = r;
  }
  return (v16bf)ur;
}

// ---------------------------------------------------------------------------
// Weight prep: pad to bf16 arena, optionally softplus(W)*gain (PosLinear).
// ---------------------------------------------------------------------------
__global__ void pad_w(const float* __restrict__ src, bf16_t* __restrict__ dst,
                      int rows, int cols, int kpad, int dstRows,
                      int do_sp, float gain) {
  int i = blockIdx.x * blockDim.x + threadIdx.x;
  int total = dstRows * kpad;
  if (i >= total) return;
  int r = i / kpad, k = i - r * kpad;
  float v = 0.f;
  if (r < rows && k < cols) {
    v = src[r * cols + k];
    if (do_sp) v = sp_f(v) * gain;
  }
  dst[i] = (bf16_t)v;
}

struct AffP {
  const float *an0w, *an0b, *wzb0, *ancw, *ancb, *wcb;
  const float *wczb0, *wczb1, *wczb2, *wczb3;
  const float *wcxb0, *wcxb1, *wcxb2, *wcxb3;
  const float *anw0, *anw1, *anw2, *anw3;
  const float *anb0, *anb1, *anb2, *anb3;
  const float *wzb1, *wzb2, *wzb3;
  const float *wxb0, *wxb1, *wxb2;
  const float *wccb0, *wccb1, *wccb2;
};

__global__ void prep_affine(AffP P, float* __restrict__ F) {
  int i = threadIdx.x;  // one block, 256 threads
  if (i < 64) {
    float s = __builtin_amdgcn_exp2f(LOG2E * P.an0w[i]);
    F[i] = s;
    F[64 + i] = P.wzb0[i] * s + P.an0b[i];
    float sc = __builtin_amdgcn_exp2f(LOG2E * P.ancw[i]);
    F[128 + i] = sc;
    F[192 + i] = P.wcb[i] * sc + P.ancb[i];
    F[256 + i] = P.wczb0[i] + kCZ0;
    const float* anw[3]  = {P.anw0, P.anw1, P.anw2};
    const float* anb[3]  = {P.anb0, P.anb1, P.anb2};
    const float* wzb[3]  = {P.wzb1, P.wzb2, P.wzb3};
    const float* wxb[3]  = {P.wxb0, P.wxb1, P.wxb2};
    const float* wccb[3] = {P.wccb0, P.wccb1, P.wccb2};
    for (int j = 0; j < 3; ++j) {
      float ps = __builtin_amdgcn_exp2f(LOG2E * anw[j][i]);
      float g = 1.f / (64.f * (float)(j + 1));
      F[928 + j * 64 + i] = ps;
      F[1120 + j * 64 + i] =
          (wzb[j][i] * g + wxb[j][i] + wccb[j][i]) * ps + anb[j][i];
    }
  }
  if (i < 128) F[320 + i] = P.wczb1[i] + kCZ0;
  if (i < 192) F[448 + i] = P.wczb2[i] + kCZ0;
  if (i < 256) F[640 + i] = P.wczb3[i] + kCZL;
  if (i < 8) {
    F[896 + i] = P.wcxb0[i] + 1.f;
    F[904 + i] = P.wcxb1[i] + 1.f;
    F[912 + i] = P.wcxb2[i] + 1.f;
    F[920 + i] = P.wcxb3[i] + 1.f;
  }
  if (i == 0) {
    F[1312] = __builtin_amdgcn_exp2f(LOG2E * P.anw3[0]);
    F[1313] = P.anb3[0];
  }
}

// ---------------------------------------------------------------------------
// Main fused kernel: 4 waves/block, 2x16 samples/wave, 40KB LDS per wave.
// Per-tile slab (bf16 elems): zb[16x256]@0 czb[16x256]@4096 ccb[16x64]@8192
//                             xb[16x32]@9216 pb[16x32]@9728
// ---------------------------------------------------------------------------
#define TILE_ELEMS 10240
#define WAVE_ELEMS (2 * TILE_ELEMS)

__global__ __launch_bounds__(128) void picnn_main(
    const float* __restrict__ x, const float* __restrict__ c,
    const bf16_t* __restrict__ W, const float* __restrict__ F,
    float* __restrict__ out) {
  extern __shared__ bf16_t smem[];
  const int lane = threadIdx.x & 31;
  const int wave = threadIdx.x >> 5;
  bf16_t* slab = smem + wave * WAVE_ELEMS;
  const int row0 = (blockIdx.x * 4 + wave) * 32;
  const int nl = lane & 15, hi = lane >> 4;

  bf16_t* zb[2]  = {slab, slab + TILE_ELEMS};
  bf16_t* czb[2] = {zb[0] + 4096, zb[1] + 4096};
  bf16_t* ccb[2] = {zb[0] + 8192, zb[1] + 8192};
  bf16_t* xb[2]  = {zb[0] + 9216, zb[1] + 9216};
  bf16_t* pb[2]  = {zb[0] + 9728, zb[1] + 9728};

  // Stage x and c tiles into LDS as bf16, K padded 8->32.
  for (int i = lane; i < 1024; i += 32) {
    int t = i >> 9, ii = i & 511;
    int r = ii >> 5, k = ii & 31;
    int row = row0 + t * 16 + r;
    float xv = (k < 8) ? x[row * 8 + k] : 0.f;
    float cv = (k < 8) ? c[row * 8 + k] : 0.f;
    xb[t][ii] = (bf16_t)xv;
    pb[t][ii] = (bf16_t)cv;
  }
  __syncthreads();

  // z0 = sp(AN(x Wz0^T + b)) ; cc = elu(AN(c Wc^T + b))
  {
    v16bf ax[2], ac[2];
#pragma unroll
    for (int t = 0; t < 2; ++t) {
      ax[t] = ldfragA(xb[t], 32, 0, lane);
      ac[t] = ldfragA(pb[t], 32, 0, lane);
    }
#pragma unroll
    for (int nt = 0; nt < 4; ++nt) {
      v16bf bz = ldfragB(W + OFF_WZ0 + nt * 16 * 32, 32, 0, lane);
      v16bf bc = ldfragB(W + OFF_WC + nt * 16 * 32, 32, 0, lane);
      int n = nt * 16 + nl;
      float s0 = F[n], t0 = F[64 + n], sc = F[128 + n], tc = F[192 + n];
#pragma unroll
      for (int t = 0; t < 2; ++t) {
        v8f az = {};
        az = wmma_acc(ax[t], bz, az);
        v8f au = {};
        au = wmma_acc(ac[t], bc, au);
#pragma unroll
        for (int r = 0; r < 8; ++r) {
          int m = r + hi * 8;
          zb[t][m * 256 + n] = (bf16_t)sp_f(az[r] * s0 + t0);
          ccb[t][m * 64 + n] = (bf16_t)elu_f(au[r] * sc + tc);
        }
      }
    }
  }
  __syncthreads();

  v16bf cf[2][2];
#pragma unroll
  for (int t = 0; t < 2; ++t) {
    cf[t][0] = ldfragA(ccb[t], 64, 0, lane);
    cf[t][1] = ldfragA(ccb[t], 64, 32, lane);
  }

  const int OFF_WCZ[4] = {OFF_WCZ0, OFF_WCZ1, OFF_WCZ2, OFF_WCZ3};
  const int OFF_WCX[4] = {OFF_WCX0, OFF_WCX1, OFF_WCX2, OFF_WCX3};
  const int OFF_WCC[3] = {OFF_WCC0, OFF_WCC1, OFF_WCC2};
  const int OFF_SPZ[3] = {OFF_SPZ1, OFF_SPZ2, OFF_SPZ3};
  const int OFF_WX[3]  = {OFF_WX0, OFF_WX1, OFF_WX2};
  const int CZOFF[4]   = {256, 320, 448, 640};

#pragma unroll
  for (int j = 0; j < 3; ++j) {
    const int KJ = 64 * (j + 1);
    // cz = sp(cc Wcz^T + b + shift) -> czb[:, 0..KJ)
    const bf16_t* Wcz = W + OFF_WCZ[j];
#pragma unroll
    for (int nt = 0; nt < 4 * (j + 1); ++nt) {
      v16bf b0 = ldfragB(Wcz + nt * 16 * 64, 64, 0, lane);
      v16bf b1 = ldfragB(Wcz + nt * 16 * 64, 64, 32, lane);
      int n = nt * 16 + nl;
      float tb = F[CZOFF[j] + n];
#pragma unroll
      for (int t = 0; t < 2; ++t) {
        v8f a = {};
        a = wmma_acc(cf[t][0], b0, a);
        a = wmma_acc(cf[t][1], b1, a);
#pragma unroll
        for (int r = 0; r < 8; ++r)
          czb[t][(r + hi * 8) * 256 + n] = (bf16_t)sp_f(a[r] + tb);
      }
    }
    // cx = cc Wcx^T + b + 1 ; pb <- x*cx (cols 8..31 stay zero)
    {
      const bf16_t* Wcx = W + OFF_WCX[j];
      v16bf b0 = ldfragB(Wcx, 64, 0, lane);
      v16bf b1 = ldfragB(Wcx, 64, 32, lane);
#pragma unroll
      for (int t = 0; t < 2; ++t) {
        v8f a = {};
        a = wmma_acc(cf[t][0], b0, a);
        a = wmma_acc(cf[t][1], b1, a);
        if (nl < 8) {
          float tb = F[896 + j * 8 + nl];
#pragma unroll
          for (int r = 0; r < 8; ++r) {
            int m = r + hi * 8;
            pb[t][m * 32 + nl] =
                (bf16_t)((float)xb[t][m * 32 + nl] * (a[r] + tb));
          }
        }
      }
    }
    __syncthreads();

    // pre = (z*cz) spWz^T + (x*cx) Wx^T + cc Wcc^T ; z_ = sp(AN(pre + b))
    const bf16_t* Wsp = W + OFF_SPZ[j];
    const bf16_t* Wxg = W + OFF_WX[j];
    const bf16_t* Wcc = W + OFF_WCC[j];
    v16bf apx[2];
#pragma unroll
    for (int t = 0; t < 2; ++t) apx[t] = ldfragA(pb[t], 32, 0, lane);
#pragma unroll
    for (int nt = 0; nt < 4; ++nt) {
      v8f a[2] = {{}, {}};
#pragma unroll
      for (int ks = 0; ks < 2 * (j + 1); ++ks) {
        v16bf bsp = ldfragB(Wsp + nt * 16 * KJ, KJ, ks * 32, lane);
#pragma unroll
        for (int t = 0; t < 2; ++t) {
          v16bf za = ldfragA(zb[t], 256, ks * 32, lane);
          v16bf ca = ldfragA(czb[t], 256, ks * 32, lane);
          a[t] = wmma_acc(pkmul(za, ca), bsp, a[t]);
        }
      }
      v16bf bx  = ldfragB(Wxg + nt * 16 * 32, 32, 0, lane);
      v16bf bc0 = ldfragB(Wcc + nt * 16 * 64, 64, 0, lane);
      v16bf bc1 = ldfragB(Wcc + nt * 16 * 64, 64, 32, lane);
      int n = nt * 16 + nl;
      float s = F[928 + j * 64 + n], tb = F[1120 + j * 64 + n];
#pragma unroll
      for (int t = 0; t < 2; ++t) {
        a[t] = wmma_acc(apx[t], bx, a[t]);
        a[t] = wmma_acc(cf[t][0], bc0, a[t]);
        a[t] = wmma_acc(cf[t][1], bc1, a[t]);
#pragma unroll
        for (int r = 0; r < 8; ++r)
          zb[t][(r + hi * 8) * 256 + KJ + n] = (bf16_t)sp_f(a[t][r] * s + tb);
      }
    }
    __syncthreads();
  }

  // Final layer: cz over 256 cols, cx, out = zcz spWz4^T + xcx Wx3^T, AN.
  {
    const bf16_t* Wcz = W + OFF_WCZ3;
#pragma unroll
    for (int nt = 0; nt < 16; ++nt) {
      v16bf b0 = ldfragB(Wcz + nt * 16 * 64, 64, 0, lane);
      v16bf b1 = ldfragB(Wcz + nt * 16 * 64, 64, 32, lane);
      int n = nt * 16 + nl;
      float tb = F[640 + n];
#pragma unroll
      for (int t = 0; t < 2; ++t) {
        v8f a = {};
        a = wmma_acc(cf[t][0], b0, a);
        a = wmma_acc(cf[t][1], b1, a);
#pragma unroll
        for (int r = 0; r < 8; ++r)
          czb[t][(r + hi * 8) * 256 + n] = (bf16_t)sp_f(a[r] + tb);
      }
    }
    {
      const bf16_t* Wcx = W + OFF_WCX3;
      v16bf b0 = ldfragB(Wcx, 64, 0, lane);
      v16bf b1 = ldfragB(Wcx, 64, 32, lane);
#pragma unroll
      for (int t = 0; t < 2; ++t) {
        v8f a = {};
        a = wmma_acc(cf[t][0], b0, a);
        a = wmma_acc(cf[t][1], b1, a);
        if (nl < 8) {
          float tb = F[920 + nl];
#pragma unroll
          for (int r = 0; r < 8; ++r) {
            int m = r + hi * 8;
            pb[t][m * 32 + nl] =
                (bf16_t)((float)xb[t][m * 32 + nl] * (a[r] + tb));
          }
        }
      }
    }
    __syncthreads();

    v8f a[2] = {{}, {}};
#pragma unroll
    for (int ks = 0; ks < 8; ++ks) {
      v16bf bsp = ldfragB(W + OFF_SPZ4, 256, ks * 32, lane);
#pragma unroll
      for (int t = 0; t < 2; ++t) {
        v16bf za = ldfragA(zb[t], 256, ks * 32, lane);
        v16bf ca = ldfragA(czb[t], 256, ks * 32, lane);
        a[t] = wmma_acc(pkmul(za, ca), bsp, a[t]);
      }
    }
    v16bf bx = ldfragB(W + OFF_WX3, 32, 0, lane);
#pragma unroll
    for (int t = 0; t < 2; ++t) {
      a[t] = wmma_acc(ldfragA(pb[t], 32, 0, lane), bx, a[t]);
      if (nl == 0) {
        float s = F[1312], tb = F[1313];
#pragma unroll
        for (int r = 0; r < 8; ++r)
          out[row0 + t * 16 + r + hi * 8] = a[t][r] * s + tb;
      }
    }
  }
}

// ---------------------------------------------------------------------------
extern "C" void kernel_launch(void* const* d_in, const int* in_sizes, int n_in,
                              void* d_out, int out_size, void* d_ws,
                              size_t ws_size, hipStream_t stream) {
  (void)n_in; (void)out_size; (void)ws_size;
  const float* x = (const float*)d_in[0];
  const float* c = (const float*)d_in[1];
  bf16_t* W = (bf16_t*)d_ws;
  float* F = (float*)((char*)d_ws + (size_t)BF_TOTAL * 2);

  auto pad = [&](int idx, int off, int rows, int cols, int kpad, int dstRows,
                 int sp, float gain) {
    int total = dstRows * kpad;
    pad_w<<<(total + 255) / 256, 256, 0, stream>>>(
        (const float*)d_in[idx], W + off, rows, cols, kpad, dstRows, sp, gain);
  };
  pad(2,  OFF_WZ0,  64, 8,   32,  64,  0, 1.f);         // Wz_ws[0]
  pad(18, OFF_WC,   64, 8,   32,  64,  0, 1.f);         // Wc_w
  pad(11, OFF_WX0,  64, 8,   32,  64,  0, 1.f);         // Wx_ws[0..2]
  pad(12, OFF_WX1,  64, 8,   32,  64,  0, 1.f);
  pad(13, OFF_WX2,  64, 8,   32,  64,  0, 1.f);
  pad(14, OFF_WX3,  1,  8,   32,  16,  0, 1.f);         // Wx_ws[3]
  pad(20, OFF_WCZ0, 64, 64,  64,  64,  0, 1.f);         // Wcz_ws
  pad(21, OFF_WCZ1, 128, 64, 64,  128, 0, 1.f);
  pad(22, OFF_WCZ2, 192, 64, 64,  192, 0, 1.f);
  pad(23, OFF_WCZ3, 256, 64, 64,  256, 0, 1.f);
  pad(28, OFF_WCX0, 8,  64,  64,  16,  0, 1.f);         // Wcx_ws
  pad(29, OFF_WCX1, 8,  64,  64,  16,  0, 1.f);
  pad(30, OFF_WCX2, 8,  64,  64,  16,  0, 1.f);
  pad(31, OFF_WCX3, 8,  64,  64,  16,  0, 1.f);
  pad(36, OFF_WCC0, 64, 64,  64,  64,  0, 1.f);         // Wcc_ws
  pad(37, OFF_WCC1, 64, 64,  64,  64,  0, 1.f);
  pad(38, OFF_WCC2, 64, 64,  64,  64,  0, 1.f);
  pad(3,  OFF_SPZ1, 64, 64,  64,  64,  1, 1.f / 64.f);  // softplus(Wz)*gain
  pad(4,  OFF_SPZ2, 64, 128, 128, 64,  1, 1.f / 128.f);
  pad(5,  OFF_SPZ3, 64, 192, 192, 64,  1, 1.f / 192.f);
  pad(6,  OFF_SPZ4, 1,  256, 256, 16,  1, 1.f / 256.f);

  AffP P;
  P.an0w = (const float*)d_in[42]; P.an0b = (const float*)d_in[43];
  P.wzb0 = (const float*)d_in[7];
  P.ancw = (const float*)d_in[52]; P.ancb = (const float*)d_in[53];
  P.wcb  = (const float*)d_in[19];
  P.wczb0 = (const float*)d_in[24]; P.wczb1 = (const float*)d_in[25];
  P.wczb2 = (const float*)d_in[26]; P.wczb3 = (const float*)d_in[27];
  P.wcxb0 = (const float*)d_in[32]; P.wcxb1 = (const float*)d_in[33];
  P.wcxb2 = (const float*)d_in[34]; P.wcxb3 = (const float*)d_in[35];
  P.anw0 = (const float*)d_in[44]; P.anw1 = (const float*)d_in[45];
  P.anw2 = (const float*)d_in[46]; P.anw3 = (const float*)d_in[47];
  P.anb0 = (const float*)d_in[48]; P.anb1 = (const float*)d_in[49];
  P.anb2 = (const float*)d_in[50]; P.anb3 = (const float*)d_in[51];
  P.wzb1 = (const float*)d_in[8];  P.wzb2 = (const float*)d_in[9];
  P.wzb3 = (const float*)d_in[10];
  P.wxb0 = (const float*)d_in[15]; P.wxb1 = (const float*)d_in[16];
  P.wxb2 = (const float*)d_in[17];
  P.wccb0 = (const float*)d_in[39]; P.wccb1 = (const float*)d_in[40];
  P.wccb2 = (const float*)d_in[41];
  prep_affine<<<1, 256, 0, stream>>>(P, F);

  int N = in_sizes[0] / 8;           // 262144
  int blocks = N / (32 * 4);         // 4 waves x 32 rows per block = 2048
  size_t lds = 4 * WAVE_ELEMS * sizeof(bf16_t);  // 160 KB
  picnn_main<<<blocks, 128, lds, stream>>>(x, c, W, F, (float*)d_out);
}
